// MultiHeadAttention_18519898980740
// MI455X (gfx1250) — matrix operand order repaired and checked by reading, not compiled, not run on hardware
//
#include <hip/hip_runtime.h>
#include <hip/hip_bf16.h>
#include <stdint.h>

#define DIM    2048
#define NHEADS 16
#define HDIM   128
#define BATCH  4
#define SEQ    2048
#define MTOT   (BATCH*SEQ)   // 8192 rows

typedef __bf16 v16bf __attribute__((ext_vector_type(16)));
typedef float  v8f   __attribute__((ext_vector_type(8)));
typedef unsigned int u32x2 __attribute__((ext_vector_type(2)));
typedef unsigned int u32x4 __attribute__((ext_vector_type(4)));
typedef int          v4i   __attribute__((ext_vector_type(4)));

// ---------- optional CDNA5 async global->LDS path ----------
#if defined(__has_builtin)
#if __has_builtin(__builtin_amdgcn_global_load_async_to_lds_b128) && \
    __has_builtin(__builtin_amdgcn_s_wait_asynccnt)
#define USE_ASYNC_LDS 1
#endif
#endif
#ifndef USE_ASYNC_LDS
#define USE_ASYNC_LDS 0
#endif

static __device__ inline void async_copy_b128(const void* g, void* l) {
#if USE_ASYNC_LDS
  __builtin_amdgcn_global_load_async_to_lds_b128(
      (__attribute__((address_space(1))) v4i*)(v4i*)g,
      (__attribute__((address_space(3))) v4i*)(v4i*)l, 0, 0);
#else
  (void)g; (void)l;
#endif
}
static __device__ inline void wait_async0() {
#if USE_ASYNC_LDS
  __builtin_amdgcn_s_wait_asynccnt(0);
#endif
}

// ---------- bf16 helpers (round-to-nearest-even) ----------
static __device__ inline unsigned short bfbits(float f) {
  unsigned u = __builtin_bit_cast(unsigned, f);
  u += 0x7FFFu + ((u >> 16) & 1u);
  return (unsigned short)(u >> 16);
}
static __device__ inline __bf16 f2bf(float f) {
  unsigned short h = bfbits(f);
  return __builtin_bit_cast(__bf16, h);
}
static __device__ inline float bf2f(__bf16 b) {
  unsigned u = ((unsigned)__builtin_bit_cast(unsigned short, b)) << 16;
  return __builtin_bit_cast(float, u);
}
static __device__ inline unsigned packbf(float lo, float hi) {
  return (unsigned)bfbits(lo) | ((unsigned)bfbits(hi) << 16);
}
static __device__ inline v16bf load_frag(const __bf16* p0, const __bf16* p1) {
  union { u32x4 u[2]; v16bf v; } f;
  f.u[0] = *(const u32x4*)p0;
  f.u[1] = *(const u32x4*)p1;
  return f.v;
}
static __device__ inline v16bf ones_frag() {
  union { u32x4 u[2]; v16bf v; } f;
  u32x4 o; o[0] = o[1] = o[2] = o[3] = 0x3F803F80u;   // bf16 1.0 pairs
  f.u[0] = o; f.u[1] = o;
  return f.v;
}
static __device__ inline v8f wmma_bf16(v16bf a, v16bf b, v8f c) {
  return __builtin_amdgcn_wmma_f32_16x16x32_bf16(false, a, false, b, (short)0, c, false, false);
}

// =====================================================================
// GEMM: C[m,n] = sum_k A[m,k] * W[n,k]. Block tile 128x128, K-step 32,
// double-buffered LDS. 8 waves = 4(M) x 2(N); wave tile 32x64 (2x4 WMMA).
// =====================================================================
template<int A_IS_BF16, int C_IS_BF16>
__global__ __launch_bounds__(256) void gemm_xwt(const void* __restrict__ Av,
                                                const float* __restrict__ W,
                                                void* __restrict__ Cv,
                                                int M, int N, int K) {
  __shared__ alignas(16) __bf16 sA[2][128*40];  // [m][k], k padded 32->40
  __shared__ alignas(16) __bf16 sB[2][128*40];  // [n][k]
  const int tid  = threadIdx.x;
  const int wid  = tid >> 5;
  const int lane = tid & 31;
  const int hf   = lane >> 4;
  const int l16  = lane & 15;
  const int wm   = wid & 3, wn = wid >> 2;
  const int mblk = blockIdx.x * 128, nblk = blockIdx.y * 128;

  v8f acc[2][4];
  #pragma unroll
  for (int i=0;i<2;++i)
    #pragma unroll
    for (int j=0;j<4;++j)
      #pragma unroll
      for (int e=0;e<8;++e) acc[i][j][e] = 0.f;

  u32x4 ra[4];
  u32x4 rw[4];

  auto load_tile_regs = [&](int kb) {
    if (A_IS_BF16) {
      const __bf16* A = (const __bf16*)Av;
      #pragma unroll
      for (int pass=0; pass<2; ++pass) {
        int row = (tid >> 2) + pass*64;
        int qd  = tid & 3;
        ra[pass] = *(const u32x4*)(A + (size_t)(mblk+row)*K + kb + qd*8);
      }
    } else {
      const float* A = (const float*)Av;
      #pragma unroll
      for (int pass=0; pass<4; ++pass) {
        int row = (tid >> 3) + pass*32;
        int qd  = tid & 7;
        ra[pass] = *(const u32x4*)(A + (size_t)(mblk+row)*K + kb + qd*4);
      }
    }
    #pragma unroll
    for (int pass=0; pass<4; ++pass) {
      int row = (tid >> 3) + pass*32;
      int qd  = tid & 7;
      rw[pass] = *(const u32x4*)(W + (size_t)(nblk+row)*K + kb + qd*4);
    }
  };
  auto store_tile_lds = [&](int buf) {
    if (A_IS_BF16) {
      #pragma unroll
      for (int pass=0; pass<2; ++pass) {
        int row = (tid >> 2) + pass*64;
        int qd  = tid & 3;
        *(u32x4*)&sA[buf][row*40 + qd*8] = ra[pass];
      }
    } else {
      #pragma unroll
      for (int pass=0; pass<4; ++pass) {
        int row = (tid >> 3) + pass*32;
        int qd  = tid & 7;
        u32x2 o;
        o[0] = packbf(__builtin_bit_cast(float, ra[pass][0]),
                      __builtin_bit_cast(float, ra[pass][1]));
        o[1] = packbf(__builtin_bit_cast(float, ra[pass][2]),
                      __builtin_bit_cast(float, ra[pass][3]));
        *(u32x2*)&sA[buf][row*40 + qd*4] = o;
      }
    }
    #pragma unroll
    for (int pass=0; pass<4; ++pass) {
      int row = (tid >> 3) + pass*32;
      int qd  = tid & 7;
      u32x2 o;
      o[0] = packbf(__builtin_bit_cast(float, rw[pass][0]),
                    __builtin_bit_cast(float, rw[pass][1]));
      o[1] = packbf(__builtin_bit_cast(float, rw[pass][2]),
                    __builtin_bit_cast(float, rw[pass][3]));
      *(u32x2*)&sB[buf][row*40 + qd*4] = o;
    }
  };

  load_tile_regs(0);
  store_tile_lds(0);

  const int NK = K / 32;
  for (int t = 0; t < NK; ++t) {
    const int cur = t & 1;
    __syncthreads();                       // publish buf[cur]

    if (t + 1 < NK) load_tile_regs((t+1)*32);
    if (t + 2 < NK) {                      // L2 prefetch two tiles ahead
      const char* pa = (const char*)Av +
          ((size_t)(mblk + (tid & 127)) * K + (t+2)*32) * (A_IS_BF16 ? 2 : 4);
      __builtin_prefetch(pa, 0, 1);
      __builtin_prefetch(W + (size_t)(nblk + (tid & 127))*K + (t+2)*32, 0, 1);
    }

    v16bf af[2], bfm[4];
    #pragma unroll
    for (int i=0;i<2;++i) {
      const __bf16* p = &sA[cur][(wm*32 + i*16 + l16)*40 + hf*8];
      af[i] = load_frag(p, p + 16);
    }
    #pragma unroll
    for (int j=0;j<4;++j) {
      const __bf16* p = &sB[cur][(wn*64 + j*16 + l16)*40 + hf*8];
      bfm[j] = load_frag(p, p + 16);
    }
    #pragma unroll
    for (int i=0;i<2;++i)
      #pragma unroll
      for (int j=0;j<4;++j)
        acc[i][j] = wmma_bf16(af[i], bfm[j], acc[i][j]);

    if (t + 1 < NK) store_tile_lds(cur ^ 1);
  }

  #pragma unroll
  for (int i=0;i<2;++i)
    #pragma unroll
    for (int j=0;j<4;++j)
      #pragma unroll
      for (int r=0;r<8;++r) {
        int m = mblk + wm*32 + i*16 + hf*8 + r;
        int n = nblk + wn*64 + j*16 + l16;
        float val = acc[i][j][r];
        if (C_IS_BF16) ((__bf16*)Cv)[(size_t)m*N + n] = f2bf(val);
        else           ((float*)Cv)[(size_t)m*N + n]  = val;
      }
}

// =====================================================================
// RoPE applied in-place to bf16 Q and K, layout [B*S, H*128].
// Q is additionally pre-scaled by 1/sqrt(HDIM) so the attention kernel's
// score WMMAs need no separate scaling multiplies.
// =====================================================================
__global__ void rope_kernel(__bf16* __restrict__ q, __bf16* __restrict__ k) {
  size_t t = (size_t)blockIdx.x * blockDim.x + threadIdx.x;
  const size_t total = (size_t)BATCH * SEQ * NHEADS * 64;
  if (t >= total) return;
  int    i   = (int)(t & 63);
  int    h   = (int)((t >> 6) & (NHEADS - 1));
  size_t row = t >> 10;
  int    s   = (int)(row & (SEQ - 1));
  float ang = (float)s * __expf((float)i * -0.14391156831212787f);
  float sn, c;
  __sincosf(ang, &sn, &c);
  const float qscale = 0.08838834764831845f;   // 1/sqrt(128)
  size_t base = row * DIM + (size_t)h * HDIM;
  {
    float a = bf2f(q[base + i]), b2 = bf2f(q[base + 64 + i]);
    q[base + i]      = f2bf((a * c - b2 * sn) * qscale);
    q[base + 64 + i] = f2bf((b2 * c + a * sn) * qscale);
  }
  {
    float a = bf2f(k[base + i]), b2 = bf2f(k[base + 64 + i]);
    k[base + i]      = f2bf(a * c - b2 * sn);
    k[base + 64 + i] = f2bf(b2 * c + a * sn);
  }
}

// =====================================================================
// Flash attention: block = one (b,h), 128 q rows (8 waves x 16 rows).
// KV tiles (32 rows) double-buffered in LDS. K staged with async
// global->LDS; V staged transposed via registers. Row sums of P are
// computed with an extra WMMA against an all-ones B matrix.
// =====================================================================
__global__ __launch_bounds__(256) void flash_attn(const __bf16* __restrict__ q,
                                                  const __bf16* __restrict__ k,
                                                  const __bf16* __restrict__ v,
                                                  __bf16* __restrict__ o) {
  __shared__ alignas(16) __bf16 sK [2][32*136];   // [kv][d] padded 128->136
  __shared__ alignas(16) __bf16 sVt[2][128*40];   // [d][kv] padded 32->40
  __shared__ alignas(16) __bf16 sP [8*16*40];     // per-wave P tile

  const int tid  = threadIdx.x;
  const int wid  = tid >> 5;
  const int lane = tid & 31;
  const int hf   = lane >> 4;
  const int l16  = lane & 15;
  const int qblk = blockIdx.x;
  const int b    = blockIdx.y >> 4;
  const int h    = blockIdx.y & 15;
  const size_t bbase = (size_t)b * SEQ * DIM + (size_t)h * HDIM;
  const __bf16* qp = q + bbase;
  const __bf16* kp = k + bbase;
  const __bf16* vp = v + bbase;

  const int qrow0 = qblk*128 + wid*16;

  v16bf qf[4];
  #pragma unroll
  for (int f=0; f<4; ++f) {
    const __bf16* p = qp + (size_t)(qrow0 + l16)*DIM + f*32 + hf*8;
    qf[f] = load_frag(p, p + 16);
  }
  const v16bf ones = ones_frag();

  float Mx[8], Lx[8];
  v8f accv[8];
  #pragma unroll
  for (int r=0;r<8;++r){ Mx[r] = -1e30f; Lx[r] = 0.f; }
  #pragma unroll
  for (int dt=0;dt<8;++dt)
    #pragma unroll
    for (int e=0;e<8;++e) accv[dt][e] = 0.f;

  const int kvK = tid >> 3, qdK = tid & 7;     // K tile: 32 rows x 8 chunks
  const int kvV = tid & 31, dsegV = tid >> 5;  // V tile: 32 rows x 8 d-segs

  u32x4 rk0, rk1, rv0, rv1;
  auto load_kv_regs = [&](int kv0, int buf) {
    const __bf16* gk = kp + (size_t)(kv0+kvK)*DIM + qdK*16;
#if USE_ASYNC_LDS
    async_copy_b128(gk,     &sK[buf][kvK*136 + qdK*16]);
    async_copy_b128(gk + 8, &sK[buf][kvK*136 + qdK*16 + 8]);
#else
    rk0 = *(const u32x4*)gk;
    rk1 = *(const u32x4*)(gk + 8);
#endif
    const __bf16* gv = vp + (size_t)(kv0+kvV)*DIM + dsegV*16;
    rv0 = *(const u32x4*)gv;
    rv1 = *(const u32x4*)(gv + 8);
    (void)buf;
  };
  auto store_kv_lds = [&](int buf) {
#if !USE_ASYNC_LDS
    *(u32x4*)&sK[buf][kvK*136 + qdK*16]     = rk0;
    *(u32x4*)&sK[buf][kvK*136 + qdK*16 + 8] = rk1;
#endif
    union { u32x4 u[2]; __bf16 e[16]; } tv;
    tv.u[0] = rv0; tv.u[1] = rv1;
    #pragma unroll
    for (int j=0;j<16;++j) sVt[buf][(dsegV*16 + j)*40 + kvV] = tv.e[j];
  };

  load_kv_regs(0, 0);
  store_kv_lds(0);

  const int NT = SEQ / 32;
  for (int kt = 0; kt < NT; ++kt) {
    const int cur = kt & 1;
    wait_async0();
    __syncthreads();                          // publish buf[cur]

    if (kt + 1 < NT) load_kv_regs((kt+1)*32, cur ^ 1);

    // scores (Q pre-scaled by 1/sqrt(d)): two 16x16 tiles, 4 WMMAs each
    v8f sc[2];
    #pragma unroll
    for (int t2=0;t2<2;++t2){
      #pragma unroll
      for (int e=0;e<8;++e) sc[t2][e] = 0.f;
      #pragma unroll
      for (int f=0;f<4;++f){
        const __bf16* p = &sK[cur][(t2*16 + l16)*136 + f*32 + hf*8];
        v16bf kf = load_frag(p, p + 16);
        sc[t2] = wmma_bf16(qf[f], kf, sc[t2]);
      }
    }

    // online softmax: shuffle reduction for row max only
    float corr[8];
    #pragma unroll
    for (int r=0;r<8;++r){
      float x0 = sc[0][r], x1 = sc[1][r];
      float m = fmaxf(x0, x1);
      m = fmaxf(m, __shfl_xor(m, 1, 32));
      m = fmaxf(m, __shfl_xor(m, 2, 32));
      m = fmaxf(m, __shfl_xor(m, 4, 32));
      m = fmaxf(m, __shfl_xor(m, 8, 32));
      float nM = fmaxf(Mx[r], m);
      corr[r]  = __expf(Mx[r] - nM);
      float p0 = __expf(x0 - nM), p1 = __expf(x1 - nM);
      sP[wid*640 + (hf*8+r)*40 + l16]      = f2bf(p0);
      sP[wid*640 + (hf*8+r)*40 + 16 + l16] = f2bf(p1);
      Mx[r] = nM;
    }

    // P as A-fragment (wave-private LDS round-trip, same-wave in-order)
    const __bf16* pp = &sP[wid*640 + l16*40 + hf*8];
    v16bf pf = load_frag(pp, pp + 16);

    // row sums via WMMA against all-ones B: rs[r] = sum_kv P[row,kv]
    v8f zero8;
    #pragma unroll
    for (int e=0;e<8;++e) zero8[e] = 0.f;
    v8f rs = wmma_bf16(pf, ones, zero8);
    #pragma unroll
    for (int r=0;r<8;++r) Lx[r] = Lx[r]*corr[r] + rs[r];

    // O += P.V over 8 d-tiles, rescaling running accumulators
    #pragma unroll
    for (int dt=0;dt<8;++dt){
      #pragma unroll
      for (int r=0;r<8;++r) accv[dt][r] *= corr[r];
      const __bf16* vb = &sVt[cur][(dt*16 + l16)*40 + hf*8];
      v16bf vf = load_frag(vb, vb + 16);
      accv[dt] = wmma_bf16(pf, vf, accv[dt]);
    }

    if (kt + 1 < NT) store_kv_lds(cur ^ 1);
  }

  // epilogue: normalize (reciprocal-multiply) and store bf16 in [B*S, DIM]
  float inv[8];
  #pragma unroll
  for (int r=0;r<8;++r) inv[r] = 1.0f / Lx[r];
  #pragma unroll
  for (int dt=0;dt<8;++dt)
    #pragma unroll
    for (int r=0;r<8;++r){
      int row = qrow0 + hf*8 + r;
      int col = h*HDIM + dt*16 + l16;
      o[((size_t)b*SEQ + row)*DIM + col] = f2bf(accv[dt][r] * inv[r]);
    }
}

// =====================================================================
extern "C" void kernel_launch(void* const* d_in, const int* in_sizes, int n_in,
                              void* d_out, int out_size, void* d_ws, size_t ws_size,
                              hipStream_t stream) {
  (void)in_sizes; (void)n_in; (void)out_size; (void)ws_size;
  const float* x  = (const float*)d_in[0];
  const float* Wq = (const float*)d_in[1];
  const float* Wk = (const float*)d_in[2];
  const float* Wv = (const float*)d_in[3];
  const float* Wo = (const float*)d_in[4];
  float* out = (float*)d_out;

  const size_t elems = (size_t)MTOT * DIM;
  __bf16* qb = (__bf16*)d_ws;
  __bf16* kb = qb + elems;
  __bf16* vb = kb + elems;
  __bf16* ab = vb + elems;

  dim3 ggrid(MTOT/128, DIM/128);
  gemm_xwt<0,1><<<ggrid, 256, 0, stream>>>(x, Wq, qb, MTOT, DIM, DIM);
  gemm_xwt<0,1><<<ggrid, 256, 0, stream>>>(x, Wk, kb, MTOT, DIM, DIM);
  gemm_xwt<0,1><<<ggrid, 256, 0, stream>>>(x, Wv, vb, MTOT, DIM, DIM);

  const size_t nrope = (size_t)BATCH * SEQ * NHEADS * 64;
  rope_kernel<<<(unsigned)((nrope + 255)/256), 256, 0, stream>>>(qb, kb);

  flash_attn<<<dim3(SEQ/128, BATCH*NHEADS), 256, 0, stream>>>(qb, kb, vb, ab);

  gemm_xwt<1,0><<<ggrid, 256, 0, stream>>>(ab, Wo, out, MTOT, DIM, DIM);
}
